// DCW_Block_55190329754128
// MI455X (gfx1250) — compile-verified, matrix operand order checked by reading, not run on hardware
//
#include <hip/hip_runtime.h>
#include <math.h>

#define Bn 4
#define Cn 64
#define Hn 96
#define Wn 320
#define RCn 16
#define Dn 48
#define HWn (Hn*Wn)      // 30720
#define BHWn (Bn*HWn)    // 122880

typedef __attribute__((ext_vector_type(16))) _Float16 v16h;
typedef __attribute__((ext_vector_type(8)))  float    v8f;

__device__ __forceinline__ float eluf(float x){ return x > 0.f ? x : expm1f(x); }
__device__ __forceinline__ float sigm(float x){ return 1.f/(1.f+expf(-x)); }
__device__ __forceinline__ int refl(int i, int n){ i = i < 0 ? -i : i; return i >= n ? 2*n-2-i : i; }

// ---------------- stage 1: q/k 1x1 convs (C=64 -> RC=16) ----------------
__global__ void k_qk(const float* __restrict__ lf, const float* __restrict__ rf,
                     const float* __restrict__ qw, const float* __restrict__ kw,
                     float* __restrict__ q, float* __restrict__ k){
  int p = blockIdx.x*blockDim.x + threadIdx.x;
  if (p >= BHWn) return;
  int b = p / HWn, hw = p % HWn;
  float aq[RCn], ak[RCn];
  #pragma unroll
  for (int o=0;o<RCn;++o){ aq[o]=0.f; ak[o]=0.f; }
  const float* lb = lf + (size_t)b*Cn*HWn + hw;
  const float* rb = rf + (size_t)b*Cn*HWn + hw;
  for (int c=0;c<Cn;++c){
    float lv = lb[(size_t)c*HWn], rv = rb[(size_t)c*HWn];
    #pragma unroll
    for (int o=0;o<RCn;++o){ aq[o] = fmaf(qw[o*Cn+c], lv, aq[o]); ak[o] = fmaf(kw[o*Cn+c], rv, ak[o]); }
  }
  #pragma unroll
  for (int o=0;o<RCn;++o){ q[((size_t)b*RCn+o)*HWn + hw] = aq[o]; k[((size_t)b*RCn+o)*HWn + hw] = ak[o]; }
}

// ---------------- group-norm stats: grid = B*8 groups, 2ch/group ----------------
__global__ void k_gnstat(const float* __restrict__ x, float* __restrict__ stat){
  int g = blockIdx.x; int b = g >> 3; int gr = g & 7; int c0 = gr*2;
  const float* base = x + ((size_t)b*RCn + c0)*HWn;   // 2 consecutive channels
  float s=0.f, ss=0.f;
  for (int i = threadIdx.x; i < 2*HWn; i += blockDim.x){ float v = base[i]; s += v; ss += v*v; }
  __shared__ float l1[256], l2[256];
  l1[threadIdx.x]=s; l2[threadIdx.x]=ss; __syncthreads();
  for (int off=128; off>0; off>>=1){
    if (threadIdx.x<off){ l1[threadIdx.x]+=l1[threadIdx.x+off]; l2[threadIdx.x]+=l2[threadIdx.x+off]; }
    __syncthreads();
  }
  if (threadIdx.x==0){
    float n = (float)(2*HWn); float m = l1[0]/n;
    stat[2*g] = m; stat[2*g+1] = l2[0]/n - m*m;
  }
}

__global__ void k_gnapply(float* __restrict__ x, const float* __restrict__ stat,
                          const float* __restrict__ g, const float* __restrict__ bb){
  int p = blockIdx.x*blockDim.x + threadIdx.x;
  if (p >= Bn*RCn*HWn) return;
  int b = p / (RCn*HWn); int r = p % (RCn*HWn); int c = r / HWn;
  int gi = b*8 + (c>>1);
  float m = stat[2*gi], v = stat[2*gi+1];
  float y = (x[p]-m)*rsqrtf(v+1e-5f)*g[c] + bb[c];
  x[p] = eluf(y);
}

// ---------------- gp = mean(q) over HW ----------------
__global__ void k_gp(const float* __restrict__ q, float* __restrict__ gp){
  int bc = blockIdx.x; // B*RC = 64
  const float* base = q + (size_t)bc*HWn;
  float s=0.f;
  for (int i=threadIdx.x;i<HWn;i+=blockDim.x) s += base[i];
  __shared__ float l[256]; l[threadIdx.x]=s; __syncthreads();
  for (int off=128;off>0;off>>=1){ if(threadIdx.x<off) l[threadIdx.x]+=l[threadIdx.x+off]; __syncthreads(); }
  if (!threadIdx.x) gp[bc] = l[0]/(float)HWn;
}

__global__ void k_gfeat(const float* gp, const float* w1, const float* b1,
                        const float* w2, const float* b2, float* gf){
  int t = threadIdx.x; if (t >= Bn*RCn) return;
  int b = t / RCn, oc = t % RCn;
  float h[8];
  #pragma unroll
  for (int j=0;j<8;++j){
    float a=b1[j];
    #pragma unroll
    for (int i=0;i<RCn;++i) a = fmaf(w1[j*RCn+i], gp[b*RCn+i], a);
    h[j]=eluf(a);
  }
  float a = b2[oc];
  #pragma unroll
  for (int j=0;j<8;++j) a = fmaf(w2[oc*8+j], h[j], a);
  gf[t] = a;
}

// ---------------- la 3x3 conv (16 -> 16, zero pad) ----------------
__global__ void k_laconv(const float* __restrict__ q, const float* __restrict__ w,
                         const float* __restrict__ bias, float* __restrict__ out){
  int p = blockIdx.x*blockDim.x + threadIdx.x; if (p>=BHWn) return;
  int b = p/HWn, hw = p%HWn, h = hw/Wn, x = hw%Wn;
  float acc[RCn];
  #pragma unroll
  for (int o=0;o<RCn;++o) acc[o]=bias[o];
  for (int kh=0;kh<3;++kh){ int hh=h+kh-1; if (hh<0||hh>=Hn) continue;
    for (int kw=0;kw<3;++kw){ int ww=x+kw-1; if (ww<0||ww>=Wn) continue;
      const float* qb = q + (size_t)b*RCn*HWn + hh*Wn + ww;
      for (int ic=0;ic<RCn;++ic){ float v = qb[(size_t)ic*HWn];
        #pragma unroll
        for (int o=0;o<RCn;++o) acc[o] = fmaf(w[(o*RCn+ic)*9 + kh*3+kw], v, acc[o]);
      }
    }
  }
  #pragma unroll
  for (int o=0;o<RCn;++o) out[((size_t)b*RCn+o)*HWn + hw] = acc[o];
}

// ---------------- attention MLP: [gfeat|lfeat](32) -> 16 -> 48 ----------------
__global__ void k_attn(const float* __restrict__ gf, const float* __restrict__ lfeat,
                       const float* __restrict__ w1, const float* __restrict__ b1,
                       const float* __restrict__ w2, const float* __restrict__ b2,
                       float* __restrict__ attn){
  int p = blockIdx.x*blockDim.x + threadIdx.x; if (p>=BHWn) return;
  int b = p/HWn, hw = p%HWn;
  float comb[32];
  #pragma unroll
  for (int i=0;i<16;++i) comb[i] = gf[b*16+i];
  #pragma unroll
  for (int i=0;i<16;++i) comb[16+i] = lfeat[((size_t)b*16+i)*HWn + hw];
  float t1[16];
  #pragma unroll
  for (int o=0;o<16;++o){
    float a=b1[o];
    #pragma unroll
    for (int i=0;i<32;++i) a = fmaf(w1[o*32+i], comb[i], a);
    t1[o]=eluf(a);
  }
  for (int d=0;d<Dn;++d){
    float a=b2[d];
    #pragma unroll
    for (int i=0;i<16;++i) a = fmaf(w2[d*16+i], t1[i], a);
    attn[((size_t)b*Dn+d)*HWn + hw] = a;
  }
}

// ---------------- km = mean over channels of k ----------------
__global__ void k_km(const float* __restrict__ k, float* __restrict__ km){
  int p = blockIdx.x*blockDim.x + threadIdx.x; if (p>=BHWn) return;
  int b = p/HWn, hw = p%HWn;
  float s=0.f;
  #pragma unroll
  for (int c=0;c<RCn;++c) s += k[((size_t)b*RCn+c)*HWn + hw];
  km[p] = s * (1.f/(float)RCn);
}

// ---------------- warp + vol = warped * sigmoid(attn), in place ----------------
__global__ void k_warpvol(const float* __restrict__ km, const float* __restrict__ directs,
                          float* __restrict__ vol){
  int p = blockIdx.x*blockDim.x + threadIdx.x; if (p >= Bn*Dn*HWn) return;
  int b = p/(Dn*HWn); int r = p%(Dn*HWn); int d = r/HWn; int hw = r%HWn;
  int h = hw/Wn; int w = hw%Wn;
  float shift = ((float)d/160.f) * directs[b] * (float)(Wn-1);
  float pos = (float)w + shift;
  float x0 = floorf(pos);
  float f  = pos - x0;
  int i0 = (int)fminf(fmaxf(x0,       0.f), (float)(Wn-1));
  int i1 = (int)fminf(fmaxf(x0 + 1.f, 0.f), (float)(Wn-1));
  const float* kb = km + (size_t)b*HWn + h*Wn;
  float wv = kb[i0]*(1.f-f) + kb[i1]*f;
  vol[p] = wv * sigm(vol[p]);
}

// ---------------- depthwise 3x3 + bias + elu ----------------
__global__ void k_dw(const float* __restrict__ vol, const float* __restrict__ w,
                     const float* __restrict__ bias, float* __restrict__ out){
  int p = blockIdx.x*blockDim.x + threadIdx.x; if (p >= Bn*Dn*HWn) return;
  int bd = p/HWn; int hw = p%HWn; int h = hw/Wn; int x = hw%Wn; int d = bd % Dn;
  float a = bias[d];
  const float* vb = vol + (size_t)bd*HWn;
  #pragma unroll
  for (int kh=0;kh<3;++kh){ int hh=h+kh-1; if (hh<0||hh>=Hn) continue;
    #pragma unroll
    for (int kw=0;kw<3;++kw){ int ww=x+kw-1; if (ww<0||ww>=Wn) continue;
      a = fmaf(w[d*9+kh*3+kw], vb[hh*Wn+ww], a);
    }
  }
  out[p] = eluf(a);
}

// ---------------- rpw 1x1 (48->48) + softmax over D, fused ----------------
__global__ void k_costsm(const float* __restrict__ dw, const float* __restrict__ w,
                         const float* __restrict__ bias, float* __restrict__ nc){
  int p = blockIdx.x*blockDim.x + threadIdx.x; if (p>=BHWn) return;
  int b = p/HWn, hw = p%HWn;
  float in[Dn];
  #pragma unroll
  for (int c=0;c<Dn;++c) in[c] = dw[((size_t)b*Dn+c)*HWn + hw];
  float cst[Dn]; float mx = -1e30f;
  #pragma unroll 4
  for (int o=0;o<Dn;++o){
    float a = bias[o];
    #pragma unroll
    for (int c=0;c<Dn;++c) a = fmaf(w[o*Dn+c], in[c], a);
    cst[o]=a; mx = fmaxf(mx, a);
  }
  float s = 0.f;
  #pragma unroll
  for (int o=0;o<Dn;++o){ float e = expf(cst[o]-mx); cst[o]=e; s+=e; }
  float inv = 1.f/s;
  #pragma unroll
  for (int o=0;o<Dn;++o) nc[((size_t)b*Dn+o)*HWn + hw] = cst[o]*inv;
}

// ---------------- weight prep: fu_w (64,112,3,3) -> f16 [tap][oc][128] padded ----------------
// fuwh2[(t*64+n)*128 + c] = fu_w[(n*112+c)*9 + t]  (c<112), else 0
__global__ void k_wprep(const float* __restrict__ w, _Float16* __restrict__ o){
  int i = blockIdx.x*blockDim.x + threadIdx.x;
  if (i >= 9*64*128) return;
  int t = i / (64*128); int r = i % (64*128); int n = r / 128; int c = r % 128;
  o[i] = (c < 112) ? (_Float16)w[(n*112 + c)*9 + t] : (_Float16)0.f;
}

// ---------------- fused 3x3 conv (112 -> 64, reflect pad) via WMMA ----------------
// Implicit GEMM: M = 64 w-positions/block, N = 64 oc, K = 9 taps x 128 (112 + 16 zero pad).
// 4 waves; wave w owns M rows 16w..16w+15, 4 N-subtile f32 accumulators.
// B tile (16KB, contiguous after k_wprep) staged with global_load_async_to_lds_b128
// (ASYNCcnt), overlapping the f32->f16 A-tile conversion. Per k-step: A frag + all
// 4 B frags loaded to registers first, then 4 back-to-back v_wmma_f32_16x16x32_f16.
__global__ void __launch_bounds__(128)
k_fused(const float* __restrict__ lf, const float* __restrict__ nc,
        const _Float16* __restrict__ wh2, const float* __restrict__ bias,
        float* __restrict__ out){
  __shared__ _Float16 Als[64][128];   // [m][k] activations
  __shared__ _Float16 Bls[64][128];   // [n][k] weights for current tap
  int tid = threadIdx.x; int lane = tid & 31; int wave = tid >> 5;
  int bx = blockIdx.x;
  int w0 = (bx % 5) * 64;
  int h  = (bx / 5) % Hn;
  int b  = bx / (5*Hn);

  // zero the A K-padding (channels 112..127) once; Bls padding comes pre-zeroed from k_wprep
  for (int i = tid; i < 64*16; i += 128)
    Als[i>>4][112 + (i&15)] = (_Float16)0.f;

  v8f zero8 = {0.f,0.f,0.f,0.f,0.f,0.f,0.f,0.f};
  v8f acc[4];
  #pragma unroll
  for (int nt=0;nt<4;++nt) acc[nt] = zero8;

  int mrow = wave*16 + (lane & 15);
  int khi  = (lane >> 4) << 3;          // 0 or 8 (lane-half K offset)
  unsigned ldsB = (unsigned)(size_t)(&Bls[0][0]);

  for (int t = 0; t < 9; ++t){
    int kh = t / 3, kw = t % 3;
    int hh = refl(h + kh - 1, Hn);
    if (t < 8){
      int hh2 = refl(h + (t+1)/3 - 1, Hn);
      __builtin_prefetch(&lf[(((size_t)b*64)*Hn + hh2)*Wn + w0], 0, 1); // global_prefetch_b8
    }
    __syncthreads();   // previous compute done before restaging

    // stage B asynchronously: contiguous 16KB tap slab -> LDS (ASYNCcnt-tracked)
    {
      const _Float16* wt = wh2 + (size_t)t * (64*128);
      #pragma unroll
      for (int i = 0; i < 8; ++i){
        unsigned off = (unsigned)((i*128 + tid) * 16);   // 128 lanes x 16B x 8 = 16KB
        asm volatile("global_load_async_to_lds_b128 %0, %1, %2"
                     :: "v"(ldsB + off), "v"(off), "s"(wt)
                     : "memory");
      }
    }

    // stage A (VALU): 64 positions x 112 channels (left_feat(64) | norm_cost(48)) -> f16
    for (int i = tid; i < 64*112; i += 128){
      int m = i / 112, c = i % 112;
      int ww = refl(w0 + m + kw - 1, Wn);
      float v = (c < 64)
        ? lf[(((size_t)b*64 + c)*Hn + hh)*Wn + ww]
        : nc[(((size_t)b*48 + (c-64))*Hn + hh)*Wn + ww];
      Als[m][c] = (_Float16)v;
    }

    asm volatile("s_wait_asynccnt 0x0" ::: "memory");  // B slab landed in LDS
    __syncthreads();

    #pragma unroll
    for (int ks = 0; ks < 4; ++ks){
      int kb = ks*32;
      union { v16h v; unsigned u[8]; } afr;
      union { v16h v; unsigned u[8]; } bfr[4];
      #pragma unroll
      for (int pq = 0; pq < 8; ++pq){
        int k0 = kb + ((pq & 4) ? 16 : 0) + khi + (pq & 3)*2;
        afr.u[pq] = *(const unsigned*)&Als[mrow][k0];
        #pragma unroll
        for (int nt = 0; nt < 4; ++nt)
          bfr[nt].u[pq] = *(const unsigned*)&Bls[nt*16 + (lane & 15)][k0];
      }
      #pragma unroll
      for (int nt = 0; nt < 4; ++nt)
        acc[nt] = __builtin_amdgcn_wmma_f32_16x16x32_f16(
            false, afr.v, false, bfr[nt].v, (short)0, acc[nt], false, false);
    }
  }
  __syncthreads();
  // C/D layout: VGPR r -> M = r + 8*(lane>=16); lane%16 -> N column
  #pragma unroll
  for (int nt=0;nt<4;++nt){
    int n = nt*16 + (lane & 15);
    float bv = bias[n];
    #pragma unroll
    for (int r=0;r<8;++r){
      int m = wave*16 + r + ((lane >> 4) << 3);
      out[(((size_t)b*64 + n)*Hn + h)*Wn + (w0 + m)] = acc[nt][r] + bv;
    }
  }
}

// ---------------- channel means of fused ----------------
__global__ void k_fmean(const float* __restrict__ f, float* __restrict__ fm){
  int bc = blockIdx.x; // B*C = 256
  const float* base = f + (size_t)bc*HWn;
  float s=0.f;
  for (int i=threadIdx.x;i<HWn;i+=blockDim.x) s+=base[i];
  __shared__ float l[256]; l[threadIdx.x]=s; __syncthreads();
  for (int off=128;off>0;off>>=1){ if(threadIdx.x<off) l[threadIdx.x]+=l[threadIdx.x+off]; __syncthreads(); }
  if (!threadIdx.x) fm[bc]=l[0]/(float)HWn;
}

// ---------------- SE: 64 -> 4 (relu) -> 64 (sigmoid) ----------------
__global__ void k_se(const float* fm, const float* w1, const float* w2, float* y){
  int t = threadIdx.x; if (t >= Bn*Cn) return;
  int b = t / Cn, c = t % Cn;
  float h[4];
  #pragma unroll
  for (int j=0;j<4;++j){
    float a=0.f;
    #pragma unroll
    for (int i=0;i<Cn;++i) a = fmaf(w1[j*Cn+i], fm[b*Cn+i], a);
    h[j] = fmaxf(a, 0.f);
  }
  float a=0.f;
  #pragma unroll
  for (int j=0;j<4;++j) a = fmaf(w2[c*4+j], h[j], a);
  y[t] = sigm(a);
}

// ---------------- x = elu(fused * y) ----------------
__global__ void k_final(const float* __restrict__ f, const float* __restrict__ y,
                        float* __restrict__ out){
  int p = blockIdx.x*blockDim.x + threadIdx.x; if (p >= Bn*Cn*HWn) return;
  int bc = p / HWn;
  out[p] = eluf(f[p] * y[bc]);
}

extern "C" void kernel_launch(void* const* d_in, const int* in_sizes, int n_in,
                              void* d_out, int out_size, void* d_ws, size_t ws_size,
                              hipStream_t stream){
  const float* lf      = (const float*)d_in[0];
  const float* rf      = (const float*)d_in[1];
  const float* directs = (const float*)d_in[2];
  const float* qw      = (const float*)d_in[3];
  const float* q_g     = (const float*)d_in[4];
  const float* q_b     = (const float*)d_in[5];
  const float* kw      = (const float*)d_in[6];
  const float* k_g     = (const float*)d_in[7];
  const float* k_b     = (const float*)d_in[8];
  const float* ga_w1   = (const float*)d_in[9];
  const float* ga_b1   = (const float*)d_in[10];
  const float* ga_w2   = (const float*)d_in[11];
  const float* ga_b2   = (const float*)d_in[12];
  const float* la_w    = (const float*)d_in[13];
  const float* la_b    = (const float*)d_in[14];
  const float* la_g    = (const float*)d_in[15];
  const float* la_bb   = (const float*)d_in[16];
  const float* af_w1   = (const float*)d_in[17];
  const float* af_b1   = (const float*)d_in[18];
  const float* af_w2   = (const float*)d_in[19];
  const float* af_b2   = (const float*)d_in[20];
  const float* rdw_w   = (const float*)d_in[21];
  const float* rdw_b   = (const float*)d_in[22];
  const float* rpw_w   = (const float*)d_in[23];
  const float* rpw_b   = (const float*)d_in[24];
  const float* fu_w    = (const float*)d_in[25];
  const float* fu_b    = (const float*)d_in[26];
  const float* se_w1   = (const float*)d_in[27];
  const float* se_w2   = (const float*)d_in[28];

  float* ws    = (float*)d_ws;
  float* q     = ws;                 // B*RC*HW = 1,966,080
  float* k     = ws + 1966080;       // 1,966,080
  float* lraw  = ws + 3932160;       // 1,966,080 (GN'd + elu'd in place -> lfeat)
  float* vol   = ws + 5898240;       // B*D*HW = 5,898,240 (attn, then vol in place)
  float* dwb   = ws + 11796480;      // 5,898,240
  float* fused = ws + 17694720;      // B*C*HW = 7,864,320
  float* sm    = ws + 25559040;      // small region
  float* qstat = sm;                 // 64
  float* kstat = sm + 64;            // 64
  float* lstat = sm + 128;           // 64
  float* gp    = sm + 192;           // 64
  float* gf    = sm + 256;           // 64
  float* fm    = sm + 320;           // 256
  float* yse   = sm + 576;           // 256
  float* km    = sm + 1024;          // 122,880
  _Float16* fuwh2 = (_Float16*)(sm + 123904);  // 9*64*128 halves (padded layout)

  float* xout  = (float*)d_out;
  float* ncout = xout + (size_t)Bn*Cn*HWn;     // norm_cost output region

  dim3 t256(256);
  k_qk     <<<480,   t256, 0, stream>>>(lf, rf, qw, kw, q, k);
  k_gnstat <<<32,    t256, 0, stream>>>(q, qstat);
  k_gnstat <<<32,    t256, 0, stream>>>(k, kstat);
  k_gnapply<<<7680,  t256, 0, stream>>>(q, qstat, q_g, q_b);
  k_gnapply<<<7680,  t256, 0, stream>>>(k, kstat, k_g, k_b);
  k_gp     <<<64,    t256, 0, stream>>>(q, gp);
  k_gfeat  <<<1,     64,   0, stream>>>(gp, ga_w1, ga_b1, ga_w2, ga_b2, gf);
  k_laconv <<<480,   t256, 0, stream>>>(q, la_w, la_b, lraw);
  k_gnstat <<<32,    t256, 0, stream>>>(lraw, lstat);
  k_gnapply<<<7680,  t256, 0, stream>>>(lraw, lstat, la_g, la_bb);
  k_attn   <<<480,   t256, 0, stream>>>(gf, lraw, af_w1, af_b1, af_w2, af_b2, vol);
  k_km     <<<480,   t256, 0, stream>>>(k, km);
  k_warpvol<<<23040, t256, 0, stream>>>(km, directs, vol);
  k_dw     <<<23040, t256, 0, stream>>>(vol, rdw_w, rdw_b, dwb);
  k_costsm <<<480,   t256, 0, stream>>>(dwb, rpw_w, rpw_b, ncout);
  k_wprep  <<<288,   t256, 0, stream>>>(fu_w, fuwh2);
  k_fused  <<<1920,  128,  0, stream>>>(lf, ncout, fuwh2, fu_b, fused);
  k_fmean  <<<256,   t256, 0, stream>>>(fused, fm);
  k_se     <<<1,     256,  0, stream>>>(fm, se_w1, se_w2, yse);
  k_final  <<<30720, t256, 0, stream>>>(fused, yse, xout);
}
